// lstm_26018911879814
// MI455X (gfx1250) — matrix operand order repaired and checked by reading, not compiled, not run on hardware
//
#include <hip/hip_runtime.h>
#include <hip/hip_bf16.h>
#include <cstdint>

typedef _Float16 f16;
typedef _Float16 v16h __attribute__((ext_vector_type(16)));
typedef _Float16 v8h  __attribute__((ext_vector_type(8)));
typedef float    v8f  __attribute__((ext_vector_type(8)));
typedef int      v4i  __attribute__((ext_vector_type(4)));

#define BB   32
#define SS   512
#define HH   512
#define EE   768
#define CC   3
#define NT   (BB*SS)      /* 16384 tokens */
#define G4H  (4*HH)       /* 2048 */

// ---------------- async copy to LDS (gfx1250 ASYNCcnt path) ----------------
#if __has_builtin(__builtin_amdgcn_global_load_async_to_lds_b128) && \
    __has_builtin(__builtin_amdgcn_s_wait_asynccnt)
#define ASYNC_OK 1
typedef __attribute__((address_space(1))) v4i* gp128;
typedef __attribute__((address_space(3))) v4i* lp128;
__device__ __forceinline__ void async_cp16(const void* g, void* l) {
  // low 32 bits of a generic LDS pointer are the LDS byte offset (AS3 addr)
  __builtin_amdgcn_global_load_async_to_lds_b128(
      (gp128)(uintptr_t)g, (lp128)(uint32_t)(uintptr_t)l, 0, 0);
}
#endif

// ---------------- workspace layout (bytes, all 256-aligned) ----------------
#define OFF_X16   ((size_t)0)                       /* NT*768  f16 */
#define OFF_WF16  ((size_t)25165824)                /* 2048x768 f16 (col-major) */
#define OFF_WB16  ((size_t)28311552)
#define OFF_W116  ((size_t)31457280)                /* 512x1024 f16 (col-major) */
#define OFF_ZXF   ((size_t)32505856)                /* NT*2048 f16 */
#define OFF_ZXB   ((size_t)99614720)
#define OFF_HCAT  ((size_t)166723584)               /* NT*1024 f16 */
#define OFF_HID   ((size_t)200278016)               /* NT*512  f32 */
#define OFF_HBUF  ((size_t)233832448)               /* 2dir*2pp*32*512 f16 */
#define OFF_BAR   ((size_t)233963520)               /* 64 u32 */
#define OFF_NLL   ((size_t)233963776)               /* NT f32 */

// ---------------- prep kernels ----------------
__global__ __launch_bounds__(256) void k_zero32(uint32_t* p, int n) {
  int i = blockIdx.x * 256 + threadIdx.x;
  if (i < n) p[i] = 0u;
}

__global__ __launch_bounds__(256) void k_gather(const float* __restrict__ emb,
                                                const int* __restrict__ ids,
                                                f16* __restrict__ X) {
  int row = blockIdx.y;                   // token
  int e = blockIdx.x * 256 + threadIdx.x; // 0..767
  X[(size_t)row * EE + e] = (f16)emb[(size_t)ids[row] * EE + e];
}

// dst[n*K + k] = (f16)src[k*N + n]  (weights -> col-major f16, one-time)
__global__ __launch_bounds__(256)
void k_transpose16(const float* __restrict__ src, f16* __restrict__ dst,
                   int K, int N) {
  int k = blockIdx.x * 16 + (threadIdx.x & 15);
  int n = blockIdx.y * 16 + (threadIdx.x >> 4);
  if (k < K && n < N) dst[(size_t)n * K + k] = (f16)src[(size_t)k * N + n];
}

__device__ __forceinline__ v8f wmma_f16(const v16h& a, const v16h& b,
                                        const v8f& c) {
  return __builtin_amdgcn_wmma_f32_16x16x32_f16(false, a, false, b, (short)0,
                                                c, false, false);
}

// ---------------- 128x64-tile WMMA GEMM (256 thr = 8 waves, 2x2 regs) ------
// B supplied COLUMN-major [N][K] (ldbt = K) so both tiles stream to LDS as-is.
// mode 0: C = A@B + bias[n] + priorK[768+prior[m]][n], f16 out (input proj)
// mode 1: C = A@B + bias[n], f32 out (dense1)
__global__ __launch_bounds__(256)
void k_gemm128(const f16* __restrict__ A, int lda, int ktiles,
               const f16* __restrict__ Bt0, const f16* __restrict__ Bt1,
               int ldbt, void* __restrict__ out0, void* __restrict__ out1,
               int ldc, const float* __restrict__ bias0,
               const float* __restrict__ bias1, const float* __restrict__ pk0,
               const float* __restrict__ pk1, const int* __restrict__ prior,
               int mode) {
  __shared__ f16 Xs[2][128 * 40];  // A tile [row][k], stride 40 (16B aligned)
  __shared__ f16 Ws[2][64 * 40];   // B tile [col][k], stride 40

  const int z = blockIdx.z;
  const f16* Bt = z ? Bt1 : Bt0;
  const float* bias = z ? bias1 : bias0;
  const float* pk = z ? pk1 : pk0;
  const int m0 = blockIdx.y * 128;
  const int n0 = blockIdx.x * 64;
  const int tid = threadIdx.x;
  const int wave = tid >> 5, lane = tid & 31;
  const int mw = wave & 3, nw = wave >> 2;
  const int lr = lane & 15, lh = lane >> 4;

  // loader coords: 16B chunk per issue; each thread: 2 A chunks + 1 B chunk
  const int lrow = tid >> 2;        // 0..63
  const int kc = (tid & 3) * 8;     // k offset of chunk

  v8f acc00 = {}, acc01 = {}, acc10 = {}, acc11 = {};

#if defined(ASYNC_OK)
#define ISSUE(KT, BUF)                                                     \
  {                                                                        \
    async_cp16(A + (size_t)(m0 + lrow) * lda + (KT) * 32 + kc,             \
               &Xs[BUF][lrow * 40 + kc]);                                  \
    async_cp16(A + (size_t)(m0 + 64 + lrow) * lda + (KT) * 32 + kc,        \
               &Xs[BUF][(64 + lrow) * 40 + kc]);                           \
    async_cp16(Bt + (size_t)(n0 + lrow) * ldbt + (KT) * 32 + kc,           \
               &Ws[BUF][lrow * 40 + kc]);                                  \
  }
  ISSUE(0, 0);
#endif

  for (int kt = 0; kt < ktiles; ++kt) {
#if defined(ASYNC_OK)
    const int buf = kt & 1;
    if (kt + 1 < ktiles) {
      ISSUE(kt + 1, buf ^ 1);
      __builtin_amdgcn_s_wait_asynccnt(3);  // in-order: current tile landed
    } else {
      __builtin_amdgcn_s_wait_asynccnt(0);
    }
    __syncthreads();
#else
    const int buf = 0;
    v8h av0 = *(const v8h*)(A + (size_t)(m0 + lrow) * lda + kt * 32 + kc);
    v8h av1 = *(const v8h*)(A + (size_t)(m0 + 64 + lrow) * lda + kt * 32 + kc);
    v8h bv = *(const v8h*)(Bt + (size_t)(n0 + lrow) * ldbt + kt * 32 + kc);
    __syncthreads();
    *(v8h*)(&Xs[0][lrow * 40 + kc]) = av0;
    *(v8h*)(&Xs[0][(64 + lrow) * 40 + kc]) = av1;
    *(v8h*)(&Ws[0][lrow * 40 + kc]) = bv;
    __syncthreads();
#endif

    // A frags: lane<16 -> k {0..7,16..23}; lane>=16 -> k {8..15,24..31}
    v16h af0, af1, bf0, bf1;
    {
      const int arow = mw * 32 + lr;
      v8h lo = *(const v8h*)(&Xs[buf][arow * 40 + lh * 8]);
      v8h hi = *(const v8h*)(&Xs[buf][arow * 40 + lh * 8 + 16]);
#pragma unroll
      for (int j = 0; j < 8; ++j) { af0[j] = lo[j]; af0[j + 8] = hi[j]; }
    }
    {
      const int arow = mw * 32 + 16 + lr;
      v8h lo = *(const v8h*)(&Xs[buf][arow * 40 + lh * 8]);
      v8h hi = *(const v8h*)(&Xs[buf][arow * 40 + lh * 8 + 16]);
#pragma unroll
      for (int j = 0; j < 8; ++j) { af1[j] = lo[j]; af1[j + 8] = hi[j]; }
    }
    // B frags: same k order as A (lane<16 -> k {0..7,16..23}); Ws is [col][k]
    {
      const int bcol = nw * 32 + lr;
      v8h lo = *(const v8h*)(&Ws[buf][bcol * 40 + lh * 8]);
      v8h hi = *(const v8h*)(&Ws[buf][bcol * 40 + lh * 8 + 16]);
#pragma unroll
      for (int j = 0; j < 8; ++j) { bf0[j] = lo[j]; bf0[j + 8] = hi[j]; }
    }
    {
      const int bcol = nw * 32 + 16 + lr;
      v8h lo = *(const v8h*)(&Ws[buf][bcol * 40 + lh * 8]);
      v8h hi = *(const v8h*)(&Ws[buf][bcol * 40 + lh * 8 + 16]);
#pragma unroll
      for (int j = 0; j < 8; ++j) { bf1[j] = lo[j]; bf1[j + 8] = hi[j]; }
    }
    acc00 = wmma_f16(af0, bf0, acc00);
    acc01 = wmma_f16(af0, bf1, acc01);
    acc10 = wmma_f16(af1, bf0, acc10);
    acc11 = wmma_f16(af1, bf1, acc11);
#if defined(ASYNC_OK)
    __syncthreads();  // tile consumed before DMA may overwrite it
#endif
  }

#pragma unroll
  for (int rt = 0; rt < 2; ++rt) {
#pragma unroll
    for (int nt = 0; nt < 2; ++nt) {
      v8f accv = rt ? (nt ? acc11 : acc10) : (nt ? acc01 : acc00);
#pragma unroll
      for (int r = 0; r < 8; ++r) {
        int m = m0 + mw * 32 + rt * 16 + r + lh * 8;
        int n = n0 + nw * 32 + nt * 16 + lr;
        float v = accv[r] + bias[n];
        if (mode == 0) {
          v += pk[(size_t)(EE + prior[m]) * G4H + n];
          ((f16*)(z ? out1 : out0))[(size_t)m * ldc + n] = (f16)v;
        } else {
          ((float*)out0)[(size_t)m * ldc + n] = v;
        }
      }
    }
  }
}

// ---------------- persistent bidirectional LSTM recurrence -----------------
// grid (32, 2): 32 column-slices x 2 directions. R-slice lives in LDS for all
// 512 steps; h state exchanged via L2 with a monotone-phase grid barrier.
__global__ __launch_bounds__(256)
void k_lstm(const f16* __restrict__ zx_f, const f16* __restrict__ zx_b,
            const float* __restrict__ Rf, const float* __restrict__ Rb,
            f16* __restrict__ hbuf, f16* __restrict__ hcat,
            unsigned* __restrict__ bar) {
  extern __shared__ char smem[];
  f16* Rst = (f16*)smem;                                   // [64col][512k] 64KB
  f16* hs = (f16*)(smem + 65536);                          // [32][512]    32KB
  float* zs = (float*)(smem + 65536 + 32768);              // [4][32][16]   8KB
  float* cs = (float*)(smem + 65536 + 32768 + 8192);       // [32][16]      2KB

  const int dir = blockIdx.y;
  const int hc0 = blockIdx.x * 16;
  const int tid = threadIdx.x;
  const int wave = tid >> 5, lane = tid & 31;
  const int g = wave & 3, mt = wave >> 2;
  const int lr = lane & 15, lh = lane >> 4;
  const float* R = dir ? Rb : Rf;
  const f16* zx = dir ? zx_b : zx_f;
  f16* h0 = hbuf + (size_t)dir * 2 * 32 * 512;
  const unsigned NBLK = gridDim.x * gridDim.y;

  // Stage recurrent-weight slice into LDS (transposed: [col][k]), f32->f16.
  for (int i = tid; i < 512 * 64; i += 256) {
    int c64 = i & 63, k = i >> 6;
    int gg = c64 >> 4, c = c64 & 15;
    Rst[(gg * 16 + c) * 512 + k] = (f16)R[(size_t)k * G4H + gg * HH + hc0 + c];
  }
  for (int i = tid; i < 512; i += 256) cs[i] = 0.0f;
  __syncthreads();

  int p = 0;
  for (int step = 0; step < SS; ++step) {
    const int t = dir ? (SS - 1 - step) : step;
    // stage full h[32][512] (f16) from global into LDS
    const f16* hg = h0 + (size_t)p * 32 * 512;
#if defined(ASYNC_OK)
    for (int i = tid; i < 2048; i += 256)
      async_cp16(hg + (size_t)i * 8, hs + i * 8);
    __builtin_amdgcn_s_wait_asynccnt(0);
#else
    for (int i = tid; i < 2048; i += 256)
      *(v8h*)(hs + i * 8) = *(const v8h*)(hg + i * 8);
#endif
    __syncthreads();

    // wave (g, mt): z-tile [16 rows x 16 cols] of gate g = h @ R
    v8f acc = {};
    const int arow = mt * 16 + lr;
#pragma unroll 4
    for (int kt = 0; kt < 16; ++kt) {
      v8h alo = *(const v8h*)(hs + arow * 512 + kt * 32 + lh * 8);
      v8h ahi = *(const v8h*)(hs + arow * 512 + kt * 32 + lh * 8 + 16);
      v16h af;
#pragma unroll
      for (int j = 0; j < 8; ++j) { af[j] = alo[j]; af[j + 8] = ahi[j]; }
      v8h b0 = *(const v8h*)(Rst + (g * 16 + lr) * 512 + kt * 32 + lh * 8);
      v8h b1 = *(const v8h*)(Rst + (g * 16 + lr) * 512 + kt * 32 + lh * 8 + 16);
      v16h bf;
#pragma unroll
      for (int j = 0; j < 8; ++j) { bf[j] = b0[j]; bf[j + 8] = b1[j]; }
      acc = wmma_f16(af, bf, acc);
    }
    // add precomputed input projection, park gates in LDS
#pragma unroll
    for (int r = 0; r < 8; ++r) {
      int m = mt * 16 + r + lh * 8;  // batch index
      int tok = m * SS + t;
      float v = acc[r] + (float)zx[(size_t)tok * G4H + g * HH + hc0 + lr];
      zs[(g * 32 + m) * 16 + lr] = v;
    }
    __syncthreads();

    // gate math: each thread owns 2 (m, col) cells
#pragma unroll
    for (int q = 0; q < 2; ++q) {
      int idx = tid * 2 + q;
      int m = idx >> 4, c = idx & 15;
      float zi = zs[(0 * 32 + m) * 16 + c];
      float zf = zs[(1 * 32 + m) * 16 + c];
      float zg = zs[(2 * 32 + m) * 16 + c];
      float zo = zs[(3 * 32 + m) * 16 + c];
      float si = 1.f / (1.f + __expf(-zi));
      float sf = 1.f / (1.f + __expf(-zf));
      float so = 1.f / (1.f + __expf(-zo));
      float cc = sf * cs[m * 16 + c] + si * tanhf(zg);
      cs[m * 16 + c] = cc;
      f16 hh = (f16)(so * tanhf(cc));
      h0[(size_t)(p ^ 1) * 32 * 512 + m * 512 + hc0 + c] = hh;
      hcat[(size_t)(m * SS + t) * 1024 + dir * 512 + hc0 + c] = hh;
    }
    p ^= 1;
    if (step != SS - 1) {
      __threadfence();
      __syncthreads();
      if (tid == 0) {
        unsigned prev = __hip_atomic_fetch_add(&bar[0], 1u, __ATOMIC_ACQ_REL,
                                               __HIP_MEMORY_SCOPE_AGENT);
        unsigned target = NBLK * (unsigned)(step + 1);
        if (prev == target - 1u) {
          __hip_atomic_store(&bar[1], (unsigned)(step + 1), __ATOMIC_RELEASE,
                             __HIP_MEMORY_SCOPE_AGENT);
        } else {
          while (__hip_atomic_load(&bar[1], __ATOMIC_ACQUIRE,
                                   __HIP_MEMORY_SCOPE_AGENT) <
                 (unsigned)(step + 1))
            __builtin_amdgcn_s_sleep(2);
        }
      }
      __syncthreads();
    }
  }
}

// ---------------- head: K=512 -> 3 logits, softmax, per-token NLL ----------
__global__ __launch_bounds__(256)
void k_head(const float* __restrict__ hid, const float* __restrict__ w2,
            const float* __restrict__ b2, const int* __restrict__ labels,
            float* __restrict__ probs, float* __restrict__ nll) {
  int wave = threadIdx.x >> 5, lane = threadIdx.x & 31;
  int t = blockIdx.x * 8 + wave;
  const float* hp = hid + (size_t)t * 512;
  float a0 = 0.f, a1 = 0.f, a2 = 0.f;
  for (int k = lane; k < 512; k += 32) {
    float h = hp[k];
    a0 += h * w2[k * 3 + 0];
    a1 += h * w2[k * 3 + 1];
    a2 += h * w2[k * 3 + 2];
  }
#pragma unroll
  for (int off = 16; off; off >>= 1) {
    a0 += __shfl_down(a0, off);
    a1 += __shfl_down(a1, off);
    a2 += __shfl_down(a2, off);
  }
  if (lane == 0) {
    a0 += b2[0]; a1 += b2[1]; a2 += b2[2];
    float mx = fmaxf(a0, fmaxf(a1, a2));
    float e0 = __expf(a0 - mx), e1 = __expf(a1 - mx), e2 = __expf(a2 - mx);
    float s = e0 + e1 + e2, inv = 1.f / s;
    probs[t * 3 + 0] = e0 * inv;
    probs[t * 3 + 1] = e1 * inv;
    probs[t * 3 + 2] = e2 * inv;
    int lab = labels[t];
    float ll = (lab == 0) ? a0 : ((lab == 1) ? a1 : a2);
    nll[t] = -(ll - mx - __logf(s));
  }
}

__global__ __launch_bounds__(256)
void k_loss(const float* __restrict__ nll, float* __restrict__ out) {
  __shared__ float red[256];
  float s = 0.f;
  for (int i = threadIdx.x; i < NT; i += 256) s += nll[i];
  red[threadIdx.x] = s;
  __syncthreads();
  for (int w = 128; w; w >>= 1) {
    if (threadIdx.x < w) red[threadIdx.x] += red[threadIdx.x + w];
    __syncthreads();
  }
  if (threadIdx.x == 0) out[0] = red[0] / (float)BB;
}

// ---------------- launcher ----------------
extern "C" void kernel_launch(void* const* d_in, const int* in_sizes, int n_in,
                              void* d_out, int out_size, void* d_ws,
                              size_t ws_size, hipStream_t stream) {
  (void)in_sizes; (void)n_in; (void)out_size; (void)ws_size;
  const float* emb   = (const float*)d_in[0];
  const float* Kf    = (const float*)d_in[1];
  const float* Rf    = (const float*)d_in[2];
  const float* bf    = (const float*)d_in[3];
  const float* Kb    = (const float*)d_in[4];
  const float* Rb    = (const float*)d_in[5];
  const float* bb    = (const float*)d_in[6];
  const float* w1    = (const float*)d_in[7];
  const float* b1    = (const float*)d_in[8];
  const float* w2    = (const float*)d_in[9];
  const float* b2    = (const float*)d_in[10];
  const int* ids     = (const int*)d_in[11];
  const int* prior   = (const int*)d_in[12];
  const int* labels  = (const int*)d_in[13];

  char* w = (char*)d_ws;
  f16* X16   = (f16*)(w + OFF_X16);
  f16* WF16  = (f16*)(w + OFF_WF16);
  f16* WB16  = (f16*)(w + OFF_WB16);
  f16* W116  = (f16*)(w + OFF_W116);
  f16* ZXF   = (f16*)(w + OFF_ZXF);
  f16* ZXB   = (f16*)(w + OFF_ZXB);
  f16* HCAT  = (f16*)(w + OFF_HCAT);
  float* HID = (float*)(w + OFF_HID);
  f16* HBUF  = (f16*)(w + OFF_HBUF);
  unsigned* BAR = (unsigned*)(w + OFF_BAR);
  float* NLL = (float*)(w + OFF_NLL);
  float* probs = (float*)d_out;
  float* loss  = (float*)d_out + (size_t)NT * 3;

  // 0) zero h state + barrier (contiguous HBUF|BAR region)
  {
    int n = (131072 + 256) / 4;
    k_zero32<<<(n + 255) / 256, 256, 0, stream>>>((uint32_t*)HBUF, n);
  }
  // 1) gather embeddings -> f16 token matrix; weights -> col-major f16
  k_gather<<<dim3(3, NT), 256, 0, stream>>>(emb, ids, X16);
  k_transpose16<<<dim3(48, 128), 256, 0, stream>>>(Kf, WF16, EE, G4H);
  k_transpose16<<<dim3(48, 128), 256, 0, stream>>>(Kb, WB16, EE, G4H);
  k_transpose16<<<dim3(64, 32), 256, 0, stream>>>(w1, W116, 1024, HH);

  // 2) input projections (fwd + bwd in z): zx = X@K[0:768] + K[768+prior] + b
  k_gemm128<<<dim3(G4H / 64, NT / 128, 2), 256, 0, stream>>>(
      X16, EE, EE / 32, WF16, WB16, EE, ZXF, ZXB, G4H, bf, bb, Kf, Kb, prior,
      0);

  // 3) persistent bidirectional recurrence
  k_lstm<<<dim3(32, 2), 256, 65536 + 32768 + 8192 + 2048, stream>>>(
      ZXF, ZXB, Rf, Rb, HBUF, HCAT, BAR);

  // 4) dense1: hid = hcat @ w1 + b1
  k_gemm128<<<dim3(HH / 64, NT / 128, 1), 256, 0, stream>>>(
      HCAT, 1024, 1024 / 32, W116, W116, 1024, HID, HID, HH, b1, b1, nullptr,
      nullptr, nullptr, 1);

  // 5) logits/softmax/per-token NLL, then deterministic loss reduction
  k_head<<<NT / 8, 256, 0, stream>>>(HID, w2, b2, labels, probs, NLL);
  k_loss<<<1, 256, 0, stream>>>(NLL, loss);
}
